// GraphDual_52046413693062
// MI455X (gfx1250) — matrix operand order repaired and checked by reading, not compiled, not run on hardware
//
#include <hip/hip_runtime.h>
#include <hip/hip_bf16.h>

typedef __attribute__((ext_vector_type(16))) _Float16 v16h;
typedef __attribute__((ext_vector_type(8)))  float    v8f;

#define NN 4096
#define EE 65536

// ---------- small utility kernels ----------
__global__ void fill_f32(float* __restrict__ p, float v, int n) {
  int t = blockIdx.x * blockDim.x + threadIdx.x;
  if (t < n) p[t] = v;
}
__global__ void fill_u32(unsigned* __restrict__ p, unsigned v, int n) {
  int t = blockIdx.x * blockDim.x + threadIdx.x;
  if (t < n) p[t] = v;
}

// ---------- B-spline basis (degree-1, K=4, DIM=3): 8 (idx, w) pairs / edge + degree ----------
__global__ void spline_kernel(const float* __restrict__ attr, const int* __restrict__ eidx,
                              int* __restrict__ idx8, float* __restrict__ bw8,
                              float* __restrict__ deg) {
  int e = blockIdx.x * blockDim.x + threadIdx.x;
  if (e >= EE) return;
  float fr[3]; int lo[3];
#pragma unroll
  for (int d = 0; d < 3; ++d) {
    float v = attr[e * 3 + d];
    v = fminf(fmaxf(v, 0.f), 1.f) * 3.f;       // * (K-1)
    float l = fminf(floorf(v), 2.f);           // clip to K-2
    fr[d] = v - l;
    lo[d] = (int)l;
  }
#pragma unroll
  for (int s = 0; s < 8; ++s) {
    int b0 = s & 1, b1 = (s >> 1) & 1, b2 = (s >> 2) & 1;
    float w = (b0 ? fr[0] : 1.f - fr[0]) * (b1 ? fr[1] : 1.f - fr[1]) * (b2 ? fr[2] : 1.f - fr[2]);
    idx8[e * 8 + s] = (lo[0] + b0) + 4 * (lo[1] + b1) + 16 * (lo[2] + b2);
    bw8[e * 8 + s] = w;
  }
  atomicAdd(&deg[eidx[EE + e]], 1.f);
}

// ---------- invdeg[n] = 1 / max(deg[n], 1) ----------
__global__ void invdeg_kernel(const float* __restrict__ deg, float* __restrict__ invdeg) {
  int n = blockIdx.x * blockDim.x + threadIdx.x;
  if (n >= NN) return;
  float d = deg[n];
  if (d < 1.f) d = 1.f;
  invdeg[n] = 1.f / d;
}

// ---------- o = bias + x @ Wr (root term) ----------
__global__ void init_out_kernel(const float* __restrict__ x, const float* __restrict__ Wr,
                                const float* __restrict__ bias, float* __restrict__ o,
                                int Cin, int Cout, int total) {
  int t = blockIdx.x * blockDim.x + threadIdx.x;
  if (t >= total) return;
  int n = t / Cout, j = t - n * Cout;
  float acc = bias[j];
  for (int ci = 0; ci < Cin; ++ci) acc += x[(size_t)n * Cin + ci] * Wr[(size_t)ci * Cout + j];
  o[t] = acc;
}

// ---------- A[dst, idx, ci] += bw * x[src, ci]  (basis-space aggregation, f32 atomics) ----------
__global__ void edge_aggregate_kernel(const float* __restrict__ x, const int* __restrict__ eidx,
                                      const int* __restrict__ idx8, const float* __restrict__ bw8,
                                      float* __restrict__ A2, int Cin, int Cin_c, int chunk_off) {
  int t = blockIdx.x * blockDim.x + threadIdx.x;
  int total = EE * Cin_c;
  if (t >= total) return;
  int e = t / Cin_c;
  int ci = t - e * Cin_c;
  int s_ = eidx[e];
  int d_ = eidx[EE + e];
  float xs = x[(size_t)s_ * Cin + chunk_off + ci];
  float* Abase = A2 + (size_t)d_ * 64 * Cin_c + ci;
#pragma unroll
  for (int s = 0; s < 8; ++s) {
    atomicAdd(Abase + idx8[e * 8 + s] * Cin_c, bw8[e * 8 + s] * xs);
  }
}

// ---------- pack A f32 -> f16 in WMMA A-fragment order ----------
// Ah[n][kb][32]: within a 32-K block, pos = kk (contiguous K),
// so lane-half h reads 16 contiguous f16 at offset h*16 (contiguous K order used here; Wp2 uses the same).
__global__ void pack_a_kernel(const float* __restrict__ A2, _Float16* __restrict__ Ah,
                              int Kc, int total) {
  int t = blockIdx.x * blockDim.x + threadIdx.x;
  if (t >= total) return;                       // total = NN * Kc
  int n = t / Kc, kkf = t - n * Kc;
  int kb = kkf >> 5, kk = kkf & 31;
  int pos = kk;
  Ah[((size_t)n * (Kc >> 5) + kb) * 32 + pos] = (_Float16)A2[t];
}

// ---------- repack chunk weights f32 -> f16 in WMMA B-fragment order ----------
// Wp2[((kb*Cout + j)*2 + half)*16 + i] = W_f16[K = kb*32 + half*16 + i][j]
__global__ void repack_w_kernel(const float* __restrict__ W, _Float16* __restrict__ Wp2,
                                int Cin, int Cout, int Cin_c, int chunk_off, int total) {
  int t = blockIdx.x * blockDim.x + threadIdx.x;
  if (t >= total) return;                       // total = 64*Cin_c*Cout
  int i = t & 15;
  int half = (t >> 4) & 1;
  int rest = t >> 5;                            // = kb*Cout + j
  int j = rest % Cout;
  int kb = rest / Cout;
  int k = kb * 32 + half * 16 + i;              // chunk-local K = b*Cin_c + ci
  int b = k / Cin_c, ci = k - b * Cin_c;
  Wp2[t] = (_Float16)W[(size_t)(b * Cin + chunk_off + ci) * Cout + j];
}

// ---------- WMMA GEMM: o[n0:16, j0:16] += (A @ W) * invdeg ----------
// Fragments are pre-swizzled: each lane loads 32 contiguous bytes (2x b128) per fragment.
__global__ void gemm_chunk_kernel(const _Float16* __restrict__ Ah,
                                  const _Float16* __restrict__ Wp2,
                                  const float* __restrict__ invdeg, float* __restrict__ o,
                                  int Cout, int Cin_c) {
  const int lane = threadIdx.x & 31;
  const int wv   = threadIdx.x >> 5;
  const int m    = lane & 15;
  const int half = lane >> 4;
  const int n0   = blockIdx.x * 16;
  const int j0   = blockIdx.y * 64 + wv * 16;
  if (j0 >= Cout) return;                       // wave-uniform; EXEC stays all-ones
  const int KB = 2 * Cin_c;                     // K-blocks of 32; always even
  const int jc = j0 + m;
  const _Float16* aptr = Ah + (size_t)(n0 + m) * KB * 32 + half * 16;
  const _Float16* bptr = Wp2 + ((size_t)jc * 2 + half) * 16;
  const size_t bstep = (size_t)Cout * 32;
  v8f acc = {};
  for (int kb = 0; kb < KB; kb += 2) {
    __builtin_prefetch(aptr + (size_t)(kb + 2) * 32, 0, 3);
    v16h a0 = *reinterpret_cast<const v16h*>(aptr + (size_t)kb * 32);
    v16h b0 = *reinterpret_cast<const v16h*>(bptr + (size_t)kb * bstep);
    v16h a1 = *reinterpret_cast<const v16h*>(aptr + (size_t)(kb + 1) * 32);
    v16h b1 = *reinterpret_cast<const v16h*>(bptr + (size_t)(kb + 1) * bstep);
    acc = __builtin_amdgcn_wmma_f32_16x16x32_f16(false, a0, false, b0, (short)0, acc,
                                                 false, false);
    acc = __builtin_amdgcn_wmma_f32_16x16x32_f16(false, a1, false, b1, (short)0, acc,
                                                 false, false);
  }
#pragma unroll
  for (int r = 0; r < 8; ++r) {                 // D: VGPR r, lane l -> M = r + 8*(l>>4), N = l&15
    int row = n0 + r + 8 * half;
    o[(size_t)row * Cout + jc] += acc[r] * invdeg[row];
  }
}

// ---------- BatchNorm stats over ELU(o) ----------
__global__ void bn_stats_kernel(const float* __restrict__ o, float* __restrict__ mu,
                                float* __restrict__ var, int Cout) {
  __shared__ float sh[256], sh2[256];
  int j = blockIdx.x;
  float s = 0.f, s2 = 0.f;
  for (int n = threadIdx.x; n < NN; n += blockDim.x) {
    float v = o[(size_t)n * Cout + j];
    v = v > 0.f ? v : expm1f(v);
    s += v; s2 += v * v;
  }
  sh[threadIdx.x] = s; sh2[threadIdx.x] = s2;
  __syncthreads();
  for (int st = blockDim.x >> 1; st > 0; st >>= 1) {
    if ((int)threadIdx.x < st) { sh[threadIdx.x] += sh[threadIdx.x + st]; sh2[threadIdx.x] += sh2[threadIdx.x + st]; }
    __syncthreads();
  }
  if (threadIdx.x == 0) {
    float mn = sh[0] * (1.f / (float)NN);
    mu[j] = mn;
    var[j] = sh2[0] * (1.f / (float)NN) - mn * mn;
  }
}

__global__ void bn_apply_kernel(const float* __restrict__ o, const float* __restrict__ mu,
                                const float* __restrict__ var, const float* __restrict__ g,
                                const float* __restrict__ be, float* __restrict__ h,
                                int Cout, int total) {
  int t = blockIdx.x * blockDim.x + threadIdx.x;
  if (t >= total) return;
  int j = t % Cout;
  float v = o[t];
  v = v > 0.f ? v : expm1f(v);
  h[t] = (v - mu[j]) * rsqrtf(var[j] + 1e-5f) * g[j] + be[j];
}

// ---------- max-pool via order-preserving float<->u32 encoding ----------
__device__ __forceinline__ unsigned f2o(float f) {
  unsigned u = __float_as_uint(f);
  return (u & 0x80000000u) ? ~u : (u | 0x80000000u);
}
__device__ __forceinline__ float o2f(unsigned u) {
  return __uint_as_float((u & 0x80000000u) ? (u & 0x7FFFFFFFu) : ~u);
}

__global__ void pool_kernel(const float* __restrict__ h, const float* __restrict__ pos,
                            const int* __restrict__ batch, unsigned* __restrict__ pbits) {
  int t = blockIdx.x * blockDim.x + threadIdx.x;
  if (t >= NN * 512) return;
  int n = t >> 9, j = t & 511;
  int cell = (int)floorf(pos[n * 3 + 0] * (1.f / 64.f)) + 2 * (int)floorf(pos[n * 3 + 1] * (1.f / 64.f));
  cell = min(max(cell, 0), 3);
  int cl = batch[n] * 4 + cell;
  atomicMax(&pbits[cl * 512 + j], f2o(h[t]));
}

__global__ void fc1_kernel(const unsigned* __restrict__ pbits, const float* __restrict__ W1,
                           float* __restrict__ z) {
  int t = blockIdx.x * blockDim.x + threadIdx.x;
  if (t >= 8 * 512) return;
  int b = t / 512, j = t - b * 512;
  float acc = 0.f;
  for (int k = 0; k < 2048; ++k) {
    float v = o2f(pbits[b * 2048 + k]);
    if ((__float_as_uint(v) & 0x7F800000u) == 0x7F800000u) v = 0.f;  // empty cluster -> 0
    acc += v * W1[(size_t)k * 512 + j];
  }
  z[t] = acc;
}

__global__ void fc2_kernel(const float* __restrict__ z, const float* __restrict__ W2,
                           float* __restrict__ out) {
  int t = blockIdx.x * blockDim.x + threadIdx.x;
  if (t >= 80) return;
  int b = t / 10, c = t - b * 10;
  float acc = 0.f;
  for (int j = 0; j < 512; ++j) acc += z[b * 512 + j] * W2[j * 10 + c];
  out[t] = acc;
}

// ---------- host orchestration ----------
extern "C" void kernel_launch(void* const* d_in, const int* in_sizes, int n_in,
                              void* d_out, int out_size, void* d_ws, size_t ws_size,
                              hipStream_t stream) {
  (void)in_sizes; (void)n_in; (void)out_size;
  const float* x0    = (const float*)d_in[0];
  const int*   eidx  = (const int*)d_in[1];
  const float* attr  = (const float*)d_in[2];
  const float* pos   = (const float*)d_in[3];
  const int*   batch = (const int*)d_in[4];
  const float* W1    = (const float*)d_in[30];
  const float* W2    = (const float*)d_in[31];
  float* out = (float*)d_out;

  size_t off = 0;
  auto take = [&](size_t bytes) -> char* {
    char* p = (char*)d_ws + off;
    off += (bytes + 255) & ~(size_t)255;
    return p;
  };
  int*      idx8  = (int*)take((size_t)EE * 8 * 4);
  float*    bw8   = (float*)take((size_t)EE * 8 * 4);
  float*    deg   = (float*)take((size_t)NN * 4);
  float*    invdg = (float*)take((size_t)NN * 4);
  float*    obuf  = (float*)take((size_t)NN * 512 * 4);
  float*    hA    = (float*)take((size_t)NN * 512 * 4);
  float*    hB    = (float*)take((size_t)NN * 512 * 4);
  float*    mu    = (float*)take(512 * 4);
  float*    var   = (float*)take(512 * 4);
  unsigned* pbits = (unsigned*)take(32 * 512 * 4);
  float*    z     = (float*)take(8 * 512 * 4);
  int CH = 64;   // shrink Cin-chunk until A(f32) + A(f16) + W(f16) buffers fit ws
  while (CH > 1 &&
         off + (size_t)NN * 64 * CH * 4 + (size_t)NN * 64 * CH * 2 +
                 (size_t)64 * CH * 512 * 2 > ws_size)
    CH >>= 1;
  float*    Abuf = (float*)take((size_t)NN * 64 * CH * 4);
  _Float16* Ah   = (_Float16*)take((size_t)NN * 64 * CH * 2);
  _Float16* Wp   = (_Float16*)take((size_t)64 * CH * 512 * 2);

  fill_f32<<<(NN + 255) / 256, 256, 0, stream>>>(deg, 0.f, NN);
  spline_kernel<<<(EE + 255) / 256, 256, 0, stream>>>(attr, eidx, idx8, bw8, deg);
  invdeg_kernel<<<(NN + 255) / 256, 256, 0, stream>>>(deg, invdg);

  const int cins[4]  = {1, 64, 128, 256};
  const int couts[4] = {64, 128, 256, 512};
  const float* cur = x0;
  for (int i = 0; i < 4; ++i) {
    int Cin = cins[i], Cout = couts[i];
    const float* W  = (const float*)d_in[5 + 5 * i + 0];
    const float* Wr = (const float*)d_in[5 + 5 * i + 1];
    const float* bs = (const float*)d_in[5 + 5 * i + 2];
    const float* gg = (const float*)d_in[5 + 5 * i + 3];
    const float* be = (const float*)d_in[5 + 5 * i + 4];
    int total = NN * Cout;
    init_out_kernel<<<(total + 255) / 256, 256, 0, stream>>>(cur, Wr, bs, obuf, Cin, Cout, total);
    int chunk = Cin < CH ? Cin : CH;
    for (int co = 0; co < Cin; co += chunk) {
      int Kc = 64 * chunk;
      int aElems = NN * Kc;
      fill_f32<<<(aElems + 255) / 256, 256, 0, stream>>>(Abuf, 0.f, aElems);
      int et = EE * chunk;
      edge_aggregate_kernel<<<(et + 255) / 256, 256, 0, stream>>>(cur, eidx, idx8, bw8, Abuf,
                                                                  Cin, chunk, co);
      pack_a_kernel<<<(aElems + 255) / 256, 256, 0, stream>>>(Abuf, Ah, Kc, aElems);
      int wElems = Kc * Cout;
      repack_w_kernel<<<(wElems + 255) / 256, 256, 0, stream>>>(W, Wp, Cin, Cout, chunk, co,
                                                                wElems);
      dim3 g(NN / 16, (Cout + 63) / 64);
      gemm_chunk_kernel<<<g, 128, 0, stream>>>(Ah, Wp, invdg, obuf, Cout, chunk);
    }
    bn_stats_kernel<<<Cout, 256, 0, stream>>>(obuf, mu, var, Cout);
    float* hout = (i & 1) ? hB : hA;
    bn_apply_kernel<<<(total + 255) / 256, 256, 0, stream>>>(obuf, mu, var, gg, be, hout, Cout,
                                                             total);
    cur = hout;
  }

  fill_u32<<<(32 * 512 + 255) / 256, 256, 0, stream>>>(pbits, 0x007FFFFFu, 32 * 512);  // enc(-inf)
  pool_kernel<<<(NN * 512 + 255) / 256, 256, 0, stream>>>(cur, pos, batch, pbits);
  fc1_kernel<<<(8 * 512 + 255) / 256, 256, 0, stream>>>(pbits, W1, z);
  fc2_kernel<<<1, 128, 0, stream>>>(z, W2, out);
  // xm = log_softmax over a [N,1] row == exactly 0 -> block 5 eliminated analytically
  fill_f32<<<(4096 + 255) / 256, 256, 0, stream>>>(out + 80, 0.f, 4096);
}
